// PolaLinearAttention_38156489458199
// MI455X (gfx1250) — compile-verified, matrix operand order checked by reading, not run on hardware
//
#include <hip/hip_runtime.h>

// ---------------------------------------------------------------------------
// PolaLinearAttention for MI455X (gfx1250), wave32, WMMA bf16 16x16x32.
// B=8 N=4096 C=512 NH=8 HD=64 KS=5 ALPHA=4 MAXP=321 EPS=1e-6, H=W=8.
// ---------------------------------------------------------------------------

#define Bv 8
#define Nv 4096
#define Cv 512
#define NHv 8
#define HDv 64
#define MAXPv 321
#define Mrows (Bv * Nv)          // 32768 token rows

typedef __attribute__((ext_vector_type(16))) __bf16 v16bf;
typedef __attribute__((ext_vector_type(8)))  float  v8f;
typedef __attribute__((ext_vector_type(8)))  unsigned short us8;

__device__ __forceinline__ unsigned short f2bf(float f) {
    unsigned int u = __float_as_uint(f);
    u = (u + 0x7fffu + ((u >> 16) & 1u)) >> 16;
    return (unsigned short)u;
}
__device__ __forceinline__ float bf2f(unsigned short s) {
    return __uint_as_float(((unsigned int)s) << 16);
}
__device__ __forceinline__ float softplusf(float x) { return log1pf(expf(x)); }
__device__ __forceinline__ float sigmoidf_(float x) { return 1.0f / (1.0f + expf(-x)); }
__device__ __forceinline__ float safe_powf(float x, float p) {
    // x >= 0 already (post-relu). x>0 ? x^p : 0
    return (x > 0.0f) ? exp2f(p * log2f(x)) : 0.0f;
}

// Fragment from two contiguous 8 x bf16 chunks (A row-major / W row-major).
__device__ __forceinline__ v16bf ldfrag(const unsigned short* p0,
                                        const unsigned short* p1) {
    union { v16bf v; us8 h[2]; } u;
    u.h[0] = *(const us8*)p0;
    u.h[1] = *(const us8*)p1;
    return u.v;
}

__device__ __forceinline__ v8f wmma_bf16(v16bf a, v16bf b, v8f c) {
    return __builtin_amdgcn_wmma_f32_16x16x32_bf16(
        false, a, false, b, (short)0, c, false, false);
}

// K-offset within a 32-wide K tile for gathered-fragment element e (0..15):
__device__ __forceinline__ int koff16(int e, int half) {
    return (e < 8) ? (half * 8 + e) : (16 + half * 8 + (e - 8));
}

// ---------------------------------------------------------------------------
// Kernel 1: fp32 -> bf16 conversions (x, qg_w, kv_w, proj_w)
// ---------------------------------------------------------------------------
__global__ void cvt_ker(const float* __restrict__ x,
                        const float* __restrict__ qg_w,
                        const float* __restrict__ kv_w,
                        const float* __restrict__ pj_w,
                        unsigned short* __restrict__ xb,
                        unsigned short* __restrict__ qgb,
                        unsigned short* __restrict__ kvb,
                        unsigned short* __restrict__ pjb) {
    const size_t NX  = (size_t)Mrows * Cv;        // 16777216
    const size_t NQG = (size_t)2 * Cv * Cv;       // 524288
    const size_t NKV = NQG;
    const size_t NPJ = (size_t)Cv * Cv;           // 262144
    size_t i = (size_t)blockIdx.x * 256 + threadIdx.x;
    if (i < NX) { xb[i] = f2bf(x[i]); return; }
    i -= NX;
    if (i < NQG) { qgb[i] = f2bf(qg_w[i]); return; }
    i -= NQG;
    if (i < NKV) { kvb[i] = f2bf(kv_w[i]); return; }
    i -= NKV;
    if (i < NPJ) { pjb[i] = f2bf(pj_w[i]); }
}

// ---------------------------------------------------------------------------
// Kernel 2/3/9: generic WMMA GEMM  D = A(M x 512) * W^T (Ncols x 512),
// wave = 16 rows x 64 cols, fused epilogue selected by `mode`:
//   mode 0: qg  -> q path (scale, powers -> qp/qn bf16), g fp32
//   mode 1: kv  -> k path (+pos_emb, scale, powers -> kp/kn bf16), v fp32
//   mode 2: proj-> += bias, fp32 out
// ---------------------------------------------------------------------------
__global__ __launch_bounds__(256) void wgemm_ker(
    const unsigned short* __restrict__ A,
    const unsigned short* __restrict__ W,
    int mode,
    const float* __restrict__ scale_p,
    const float* __restrict__ power_p,
    const float* __restrict__ pos_emb,
    const float* __restrict__ proj_b,
    unsigned short* __restrict__ out_p,
    unsigned short* __restrict__ out_n,
    float* __restrict__ out_f) {
    const int lane = threadIdx.x & 31;
    const int w    = threadIdx.x >> 5;
    const int half = lane >> 4;
    const int nloc = lane & 15;
    const int row0 = blockIdx.x * 16;
    const int cg   = blockIdx.y * 8 + w;   // 64-col group
    const int col0 = cg * 64;

    const unsigned short* arow  = A + (size_t)(row0 + nloc) * Cv;
    const unsigned short* wbase = W + (size_t)(col0 + nloc) * Cv;

    v8f acc[4] = {};
    for (int kb = 0; kb < Cv; kb += 32) {
        v16bf a = ldfrag(arow + kb + half * 8, arow + kb + 16 + half * 8);
#pragma unroll
        for (int t = 0; t < 4; t++) {
            const unsigned short* wr = wbase + (size_t)t * 16 * Cv;
            v16bf bm = ldfrag(wr + kb + half * 8, wr + kb + 16 + half * 8);
            acc[t] = wmma_bf16(a, bm, acc[t]);
        }
    }

#pragma unroll
    for (int t = 0; t < 4; t++) {
        const int c = col0 + t * 16 + nloc;
#pragma unroll
        for (int r = 0; r < 8; r++) {
            const int m = row0 + r + 8 * half;     // global token row
            float val = acc[t][r];
            if (mode == 2) {
                out_f[(size_t)m * Cv + c] = val + proj_b[c];
            } else if (c < Cv) {
                float sc = softplusf(scale_p[c]);
                float pw = 1.0f + 4.0f * sigmoidf_(power_p[c]);
                float qk;
                if (mode == 1) {
                    int nseq = m & (Nv - 1);
                    float pe = pos_emb[(size_t)(nseq % MAXPv) * Cv + c];
                    qk = (val + pe) / sc;
                } else {
                    qk = val / sc;
                }
                out_p[(size_t)m * Cv + c] = f2bf(safe_powf(fmaxf(qk, 0.0f), pw));
                out_n[(size_t)m * Cv + c] = f2bf(safe_powf(fmaxf(-qk, 0.0f), pw));
            } else {
                out_f[(size_t)m * Cv + (c - Cv)] = val;   // g or v
            }
        }
    }
}

// ---------------------------------------------------------------------------
// Kernel 4: k_mean[b,h,d(128)] = mean_n [kp|kn]
// ---------------------------------------------------------------------------
__global__ void kmean_ker(const unsigned short* __restrict__ KP,
                          const unsigned short* __restrict__ KN,
                          float* __restrict__ KM) {
    const int bh = blockIdx.x;
    const int b = bh >> 3, h = bh & 7;
    const int d = threadIdx.x;                 // 0..127
    const unsigned short* src = (d < HDv) ? KP : KN;
    const int col = h * HDv + (d & (HDv - 1));
    float s = 0.0f;
    for (int n = 0; n < Nv; n++)
        s += bf2f(src[(size_t)(b * Nv + n) * Cv + col]);
    KM[(size_t)bh * 128 + d] = s * (1.0f / Nv);
}

// ---------------------------------------------------------------------------
// Kernel 5: KVfull[b,h,d(128),e(64)] = (1/N) sum_n kk[n,d] * v[n, h*64+e]
// One WG per (b,h); wave w owns d-tile w*16, all 64 cols (4 accs).
// ---------------------------------------------------------------------------
__global__ __launch_bounds__(256) void kvouter_ker(
    const unsigned short* __restrict__ KP,
    const unsigned short* __restrict__ KN,
    const float* __restrict__ V,
    float* __restrict__ KVF) {
    const int bh = blockIdx.x;
    const int b = bh >> 3, h = bh & 7;
    const int lane = threadIdx.x & 31;
    const int w    = threadIdx.x >> 5;
    const int half = lane >> 4, nloc = lane & 15;
    const int d0   = w * 16;
    const int drow = d0 + nloc;
    const unsigned short* ksrc = (drow < HDv) ? KP : KN;
    const int dcol = h * HDv + (drow & (HDv - 1));

    v8f acc[4] = {};
    for (int nb = 0; nb < Nv; nb += 32) {
        union { v16bf v; unsigned short s[16]; } ua;
#pragma unroll
        for (int e = 0; e < 16; e++) {
            int n = nb + koff16(e, half);
            ua.s[e] = ksrc[(size_t)(b * Nv + n) * Cv + dcol];
        }
#pragma unroll
        for (int t = 0; t < 4; t++) {
            const int eg = t * 16 + nloc;      // 0..63
            union { v16bf v; unsigned short s[16]; } ub;
#pragma unroll
            for (int e = 0; e < 16; e++) {
                int n = nb + koff16(e, half);
                ub.s[e] = f2bf(V[(size_t)(b * Nv + n) * Cv + h * HDv + eg]);
            }
            acc[t] = wmma_bf16(ua.v, ub.v, acc[t]);
        }
    }
    const float rn2 = 1.0f / (float)Nv;
#pragma unroll
    for (int t = 0; t < 4; t++)
#pragma unroll
        for (int r = 0; r < 8; r++) {
            int d = d0 + r + 8 * half;
            int e = t * 16 + nloc;
            KVF[((size_t)bh * 128 + d) * 64 + e] = acc[t][r] * rn2;
        }
}

// ---------------------------------------------------------------------------
// Kernel 6: z1/z2[b,h,n] = 1/(q_{sim,opp} . k_mean + eps)
// ---------------------------------------------------------------------------
__global__ void z_ker(const unsigned short* __restrict__ QP,
                      const unsigned short* __restrict__ QN,
                      const float* __restrict__ KM,
                      float* __restrict__ Z1, float* __restrict__ Z2) {
    size_t i = (size_t)blockIdx.x * 256 + threadIdx.x;   // (bh)*4096 + n
    if (i >= (size_t)Bv * NHv * Nv) return;
    const int bh = (int)(i >> 12);
    const int n  = (int)(i & (Nv - 1));
    const int b = bh >> 3, h = bh & 7;
    const float* km = KM + (size_t)bh * 128;
    const size_t base = (size_t)(b * Nv + n) * Cv + h * HDv;
    float d1 = 0.0f, d2 = 0.0f;
    for (int d = 0; d < HDv; d++) {
        float qp = bf2f(QP[base + d]);
        float qn = bf2f(QN[base + d]);
        d1 += qp * km[d] + qn * km[64 + d];
        d2 += qn * km[d] + qp * km[64 + d];
    }
    Z1[i] = 1.0f / (d1 + 1e-6f);
    Z2[i] = 1.0f / (d2 + 1e-6f);
}

// ---------------------------------------------------------------------------
// Kernel 7: depthwise 5x5 conv on v viewed as (B, 64, 4096, 8), scattered
// into vd laid out (B, N, C).
// ---------------------------------------------------------------------------
__global__ void dwconv_ker(const float* __restrict__ V,
                           const float* __restrict__ Wc,
                           const float* __restrict__ Bc,
                           float* __restrict__ VD) {
    size_t idx = (size_t)blockIdx.x * 256 + threadIdx.x;
    if (idx >= (size_t)Bv * 64 * Nv * 8) return;
    const int j = (int)(idx & 7);  size_t t = idx >> 3;
    const int i = (int)(t & (Nv - 1)); t >>= 12;
    const int c = (int)(t & 63);
    const int b = (int)(t >> 6);
    const int h = c >> 3;
    float sum = Bc[c];
#pragma unroll
    for (int di = 0; di < 5; di++) {
        const int ii = i + di - 2;
        if (ii < 0 || ii >= Nv) continue;
#pragma unroll
        for (int dj = 0; dj < 5; dj++) {
            const int jj = j + dj - 2;
            if (jj < 0 || jj >= 8) continue;
            const int rem = (c & 7) * 32768 + ii * 8 + jj;
            const int n = rem >> 6, d = rem & 63;
            sum += V[(size_t)(b * Nv + n) * Cv + h * HDv + d] *
                   Wc[c * 25 + di * 5 + dj];
        }
    }
    const int f  = c * 32768 + i * 8 + j;     // flat within (512, 4096)
    const int c2 = f >> 12, n2 = f & (Nv - 1);
    VD[(size_t)(b * Nv + n2) * Cv + c2] = sum;
}

// ---------------------------------------------------------------------------
// Kernel 8: per-(b,h) GEMM  xa = q_sim(N x 128) * KV'(128 x 64), with the
// q_opp half-swap folded into the B-fragment row index for cols >= 32.
// Epilogue: *z, +vd, *g -> bf16 T buffer (input to proj GEMM).
// ---------------------------------------------------------------------------
__global__ __launch_bounds__(256) void attn_ker(
    const unsigned short* __restrict__ QP,
    const unsigned short* __restrict__ QN,
    const float* __restrict__ KVF,
    const float* __restrict__ Z1, const float* __restrict__ Z2,
    const float* __restrict__ VD, const float* __restrict__ G,
    unsigned short* __restrict__ T) {
    const int bh = blockIdx.x;
    const int b = bh >> 3, h = bh & 7;
    const int lane = threadIdx.x & 31;
    const int w    = threadIdx.x >> 5;
    const int half = lane >> 4, nloc = lane & 15;
    const int cgrp = w & 1;        // 32-col group within the head
    const int qrt  = w >> 1;       // 0..3: n-tile quarter

    // Prepack B-fragments (constant over n): 4 K-tiles x 2 col-tiles.
    v16bf bfr[4][2];
#pragma unroll
    for (int kb = 0; kb < 4; kb++)
#pragma unroll
        for (int ct = 0; ct < 2; ct++) {
            const int e = cgrp * 32 + ct * 16 + nloc;   // 0..63
            union { v16bf v; unsigned short s[16]; } ub;
#pragma unroll
            for (int el = 0; el < 16; el++) {
                int k = kb * 32 + koff16(el, half);
                int row = (e < 32) ? k : ((k + 64) & 127);  // q_opp swap
                ub.s[el] = f2bf(KVF[((size_t)bh * 128 + row) * 64 + e]);
            }
            bfr[kb][ct] = ub.v;
        }

    for (int s = 0; s < 8; s++) {
        const int nt = blockIdx.y * 32 + qrt * 8 + s;    // 0..255
        const int tn = nt * 16;
        const size_t arow = (size_t)(b * Nv + tn + nloc) * Cv + h * HDv;
        v8f acc[2] = {};
#pragma unroll
        for (int kb = 0; kb < 4; kb++) {
            const int dA = kb * 32 + half * 8;
            const int dB = dA + 16;
            const unsigned short* pa =
                (dA < HDv) ? (QP + arow + dA) : (QN + arow + (dA - HDv));
            const unsigned short* pb =
                (dB < HDv) ? (QP + arow + dB) : (QN + arow + (dB - HDv));
            v16bf a = ldfrag(pa, pb);
            acc[0] = wmma_bf16(a, bfr[kb][0], acc[0]);
            acc[1] = wmma_bf16(a, bfr[kb][1], acc[1]);
        }
#pragma unroll
        for (int ct = 0; ct < 2; ct++) {
            const int ch = cgrp * 32 + ct * 16 + nloc;   // 0..63 in head
            const int cidx = h * HDv + ch;
            const float* Z = (ch < 32) ? Z1 : Z2;
#pragma unroll
            for (int r = 0; r < 8; r++) {
                const int ntok = tn + r + 8 * half;
                const float z = Z[(size_t)bh * Nv + ntok];
                const size_t gi = (size_t)(b * Nv + ntok) * Cv + cidx;
                T[gi] = f2bf((acc[ct][r] * z + VD[gi]) * G[gi]);
            }
        }
    }
}

// ---------------------------------------------------------------------------
// Workspace layout (bytes, all 256-aligned)
// ---------------------------------------------------------------------------
#define S_XB  ((size_t)Mrows * Cv * 2)
#define S_W2C ((size_t)2 * Cv * Cv * 2)
#define S_W1C ((size_t)Cv * Cv * 2)
#define S_BF  ((size_t)Mrows * Cv * 2)
#define S_F32 ((size_t)Mrows * Cv * 4)
#define O_XB   ((size_t)0)
#define O_WQG  (O_XB  + S_XB)
#define O_WKV  (O_WQG + S_W2C)
#define O_WPJ  (O_WKV + S_W2C)
#define O_QP   (O_WPJ + S_W1C)
#define O_QN   (O_QP  + S_BF)
#define O_KP   (O_QN  + S_BF)
#define O_KN   (O_KP  + S_BF)
#define O_G    (O_KN  + S_BF)
#define O_V    (O_G   + S_F32)
#define O_KM   (O_V   + S_F32)
#define O_KVF  (O_KM  + (size_t)Bv * NHv * 128 * 4)
#define O_Z1   (O_KVF + (size_t)Bv * NHv * 128 * 64 * 4)
#define O_Z2   (O_Z1  + (size_t)Bv * NHv * Nv * 4)
#define O_VD   (O_Z2  + (size_t)Bv * NHv * Nv * 4)
#define O_TB   (O_VD  + S_F32)

extern "C" void kernel_launch(void* const* d_in, const int* in_sizes, int n_in,
                              void* d_out, int out_size, void* d_ws, size_t ws_size,
                              hipStream_t stream) {
    (void)in_sizes; (void)n_in; (void)out_size; (void)ws_size;
    const float* x       = (const float*)d_in[0];
    const float* qg_w    = (const float*)d_in[1];
    const float* kv_w    = (const float*)d_in[2];
    const float* proj_w  = (const float*)d_in[3];
    const float* proj_b  = (const float*)d_in[4];
    const float* power_p = (const float*)d_in[5];
    const float* scale_p = (const float*)d_in[6];
    const float* pos_emb = (const float*)d_in[7];
    const float* dwc_w   = (const float*)d_in[8];
    const float* dwc_b   = (const float*)d_in[9];
    float* out = (float*)d_out;
    char* ws = (char*)d_ws;

    unsigned short* XB  = (unsigned short*)(ws + O_XB);
    unsigned short* WQG = (unsigned short*)(ws + O_WQG);
    unsigned short* WKV = (unsigned short*)(ws + O_WKV);
    unsigned short* WPJ = (unsigned short*)(ws + O_WPJ);
    unsigned short* QP  = (unsigned short*)(ws + O_QP);
    unsigned short* QN  = (unsigned short*)(ws + O_QN);
    unsigned short* KP  = (unsigned short*)(ws + O_KP);
    unsigned short* KN  = (unsigned short*)(ws + O_KN);
    float* Gf  = (float*)(ws + O_G);
    float* Vf  = (float*)(ws + O_V);
    float* KM  = (float*)(ws + O_KM);
    float* KVF = (float*)(ws + O_KVF);
    float* Z1  = (float*)(ws + O_Z1);
    float* Z2  = (float*)(ws + O_Z2);
    float* VD  = (float*)(ws + O_VD);
    unsigned short* TB = (unsigned short*)(ws + O_TB);

    // 1) convert x + weights to bf16
    {
        size_t total = (size_t)Mrows * Cv + 2 * (size_t)2 * Cv * Cv + (size_t)Cv * Cv;
        cvt_ker<<<dim3((unsigned)((total + 255) / 256)), 256, 0, stream>>>(
            x, qg_w, kv_w, proj_w, XB, WQG, WKV, WPJ);
    }
    // 2) qg GEMM (epilogue: q powers + g)
    wgemm_ker<<<dim3(Mrows / 16, 2), 256, 0, stream>>>(
        XB, WQG, 0, scale_p, power_p, pos_emb, proj_b, QP, QN, Gf);
    // 3) kv GEMM (epilogue: k powers (+pe) + v)
    wgemm_ker<<<dim3(Mrows / 16, 2), 256, 0, stream>>>(
        XB, WKV, 1, scale_p, power_p, pos_emb, proj_b, KP, KN, Vf);
    // 4) k_mean
    kmean_ker<<<dim3(Bv * NHv), 128, 0, stream>>>(KP, KN, KM);
    // 5) KVfull = kk^T v / N
    kvouter_ker<<<dim3(Bv * NHv), 256, 0, stream>>>(KP, KN, Vf, KVF);
    // 6) z1 / z2
    z_ker<<<dim3((Bv * NHv * Nv) / 256), 256, 0, stream>>>(QP, QN, KM, Z1, Z2);
    // 7) depthwise conv -> vd
    dwconv_ker<<<dim3((Bv * 64 * Nv * 8) / 256), 256, 0, stream>>>(
        Vf, dwc_w, dwc_b, VD);
    // 8) attention GEMM + (*z + vd) * g -> T bf16
    attn_ker<<<dim3(Bv * NHv, 8), 256, 0, stream>>>(
        QP, QN, KVF, Z1, Z2, VD, Gf, TB);
    // 9) proj GEMM + bias -> out fp32
    wgemm_ker<<<dim3(Mrows / 16, 1), 256, 0, stream>>>(
        TB, WPJ, 2, scale_p, power_p, pos_emb, proj_b, QP, QN, out);
}
